// NoteFramePredictor_72224170049929
// MI455X (gfx1250) — matrix operand order repaired and checked by reading, not compiled, not run on hardware
//
#include <hip/hip_runtime.h>

#define TDIM 6000
#define BDIM 16
#define HDIM 256
#define CDIM 105
#define MGAP 19
#define RGAP 9

typedef __attribute__((ext_vector_type(16))) __bf16         v16bf;
typedef __attribute__((ext_vector_type(8)))  float          v8f;
typedef __attribute__((ext_vector_type(4)))  float          v4f;

// f32 -> (bf16 hi, bf16 lo) split; native __bf16 casts are RNE and let the
// backend use packed bf16 converts where the ISA has them.
__device__ __forceinline__ void split_bf16(float x, __bf16& hi, __bf16& lo) {
    __bf16 h = (__bf16)x;
    hi = h;
    lo = (__bf16)(x - (float)h);
}

// ---------------------------------------------------------------------------
// One-time W pre-pack: split W (105x256 f32) into bf16 hi/lo, stored in the
// per-lane B-operand layout used here (contiguous K, same as A):  idx = ((n*8+ks)*32 + lane)*16 + e
// where lane holds column c = n*16+(lane&15) and K = ks*32 + 16*(lane>>4) + e.
// ---------------------------------------------------------------------------
__global__ __launch_bounds__(32) void pack_W_kernel(
    const float* __restrict__ Wm, __bf16* __restrict__ Whi,
    __bf16* __restrict__ Wlo)
{
    const int pair = blockIdx.x;          // 0..55 = n*8 + ks
    const int n    = pair >> 3;
    const int ks   = pair & 7;
    const int lane = threadIdx.x;
    const int hsel = lane >> 4;
    const int l16  = lane & 15;
    const int c    = n * 16 + l16;
    const int k0   = ks * 32 + 16 * hsel;

    v16bf uh, ul;
    if (c < CDIM) {
        const float* wp = Wm + (size_t)c * HDIM + k0;
#pragma unroll
        for (int e = 0; e < 16; ++e) {
            __bf16 h, l;
            split_bf16(wp[e], h, l);
            uh[e] = h; ul[e] = l;
        }
    } else {
#pragma unroll
        for (int e = 0; e < 16; ++e) { uh[e] = (__bf16)0.0f; ul[e] = (__bf16)0.0f; }
    }
    ((v16bf*)Whi)[pair * 32 + lane] = uh;
    ((v16bf*)Wlo)[pair * 32 + lane] = ul;
}

// ---------------------------------------------------------------------------
// GEMM: logits = feat(96000x256) @ W^T + b via bf16x3 WMMA.
// Each wave owns TWO 16(M) x 112(N) strips (32 rows), 8 waves/block.
// B operands come pre-packed (no conversion VALU); A is loaded non-temporally
// (streamed once) and split in-flight. LDS-staged epilogue for coalesced
// non-temporal stores of the 40MB logits.
// ---------------------------------------------------------------------------
__global__ __launch_bounds__(256) void gemm_logits_kernel(
    const float* __restrict__ feat, const __bf16* __restrict__ Whi,
    const __bf16* __restrict__ Wlo, const float* __restrict__ bias,
    float* __restrict__ out_bd, float* __restrict__ out_frame,
    float* __restrict__ probs)
{
    __shared__ float tile[8][16][112];  // 57,344 B, per-wave private slice

    const int lane = threadIdx.x & 31;
    const int wv   = threadIdx.x >> 5;
    const int mt0  = (blockIdx.x * 8 + wv) * 2;   // first of 2 m-tiles
    const int hsel = lane >> 4;
    const int l16  = lane & 15;

    const float* arp[2];
    arp[0] = feat + (size_t)(mt0 * 16 + l16) * HDIM;
    arp[1] = arp[0] + (size_t)16 * HDIM;

    v8f acc[2][7];
#pragma unroll
    for (int s = 0; s < 2; ++s)
#pragma unroll
        for (int n = 0; n < 7; ++n)
            acc[s][n] = (v8f){0.f, 0.f, 0.f, 0.f, 0.f, 0.f, 0.f, 0.f};

    const v16bf* bh_base = (const v16bf*)Whi;
    const v16bf* bl_base = (const v16bf*)Wlo;

#pragma unroll 2
    for (int ks = 0; ks < 8; ++ks) {
        const int k0 = ks * 32;

        // ---- A tiles (streamed, NT): lane<16 K={0..15}, lane>=16 K={16..31}
        v16bf a_hi[2], a_lo[2];
#pragma unroll
        for (int s = 0; s < 2; ++s) {
            const v4f* p0 = (const v4f*)(arp[s] + k0 + 16 * hsel);
            const v4f* p1 = (const v4f*)(arp[s] + k0 + 8 + 16 * hsel);
            v4f a0 = __builtin_nontemporal_load(p0);
            v4f a1 = __builtin_nontemporal_load(p0 + 1);
            v4f a2 = __builtin_nontemporal_load(p1);
            v4f a3 = __builtin_nontemporal_load(p1 + 1);
            float af[16];
#pragma unroll
            for (int e = 0; e < 4; ++e) {
                af[e] = a0[e]; af[4 + e] = a1[e]; af[8 + e] = a2[e]; af[12 + e] = a3[e];
            }
            v16bf h, l;
#pragma unroll
            for (int e = 0; e < 16; ++e) {
                __bf16 hh, ll;
                split_bf16(af[e], hh, ll);
                h[e] = hh; l[e] = ll;
            }
            a_hi[s] = h; a_lo[s] = l;
        }

        // ---- pre-packed B tiles: two 32B vector loads per (n,ks), L2-resident
#pragma unroll
        for (int n = 0; n < 7; ++n) {
            const int  bidx = (n * 8 + ks) * 32 + lane;
            const v16bf b_hi = bh_base[bidx];
            const v16bf b_lo = bl_base[bidx];
#pragma unroll
            for (int s = 0; s < 2; ++s) {
                // bf16x3: hi*hi + hi*lo + lo*hi  (~fp32 accuracy)
                acc[s][n] = __builtin_amdgcn_wmma_f32_16x16x32_bf16(
                    false, a_hi[s], false, b_hi, (short)0, acc[s][n], false, false);
                acc[s][n] = __builtin_amdgcn_wmma_f32_16x16x32_bf16(
                    false, a_hi[s], false, b_lo, (short)0, acc[s][n], false, false);
                acc[s][n] = __builtin_amdgcn_wmma_f32_16x16x32_bf16(
                    false, a_lo[s], false, b_hi, (short)0, acc[s][n], false, false);
            }
        }
    }

    // ---- epilogue: per sub-tile, add bias, stage in LDS, coalesced NT stores
    float bb[7];
#pragma unroll
    for (int n = 0; n < 7; ++n) {
        const int c = n * 16 + l16;
        bb[n] = (c < CDIM) ? bias[c] : 0.0f;
    }

    for (int s = 0; s < 2; ++s) {
#pragma unroll
        for (int n = 0; n < 7; ++n) {
            const int c = n * 16 + l16;
#pragma unroll
            for (int r = 0; r < 8; ++r)
                tile[wv][r + 8 * hsel][c] = acc[s][n][r] + bb[n];
        }
        __syncthreads();

        const int m0 = (mt0 + s) * 16;
        if (lane < 16) {  // channel 0: note_bd_logits + sigmoid probs
            float v = tile[wv][lane][0];
            v = fminf(16.0f, fmaxf(-16.0f, v));
            __builtin_nontemporal_store(v, &out_bd[m0 + lane]);
            probs[m0 + lane] = 1.0f / (1.0f + __expf(-v));
        }
        for (int idx = lane; idx < 16 * 104; idx += 32) {
            const int r  = idx / 104;
            const int cc = idx - r * 104;  // frame channel = logits col cc+1
            __builtin_nontemporal_store(tile[wv][r][cc + 1],
                                        &out_frame[(size_t)(m0 + r) * 104 + cc]);
        }
        __syncthreads();
    }
}

// ---------------------------------------------------------------------------
// Sequential boundary regulation: one wave per batch row. Lanes stage probs /
// word_bd bits into LDS coalesced; lane 0 runs the serial phase1/phase2 scan
// against LDS; lanes re-fan-out for the masked store.
// ---------------------------------------------------------------------------
__global__ __launch_bounds__(32) void boundary_scan_kernel(
    const float* __restrict__ probs, const int* __restrict__ word_bd,
    const float* __restrict__ non_padding, float* __restrict__ out_pred)
{
    __shared__ float         pr[TDIM];
    __shared__ unsigned char res[TDIM + 8];
    __shared__ unsigned int  refb[(TDIM + 31) / 32];
    __shared__ float         partial[32];
    __shared__ int           Ls;

    const int row  = blockIdx.x;
    const int lane = threadIdx.x;
    const float* prg = probs + (size_t)row * TDIM;
    const int*   wbg = word_bd + (size_t)row * TDIM;
    const float* npg = non_padding + (size_t)row * TDIM;

    float s = 0.0f;
    for (int j = lane; j < TDIM; j += 32) {
        pr[j]  = prg[j];
        res[j] = 0;
        s += npg[j];
    }
    if (lane < 8) res[TDIM + lane] = 0;
    for (int w = lane; w < (TDIM + 31) / 32; w += 32) {
        unsigned int bits = 0;
        const int base = w * 32;
        for (int k = 0; k < 32; ++k) {
            const int j = base + k;
            if (j < TDIM && wbg[j] == 1) bits |= (1u << k);
        }
        refb[w] = bits;
    }
    partial[lane] = s;
    __syncthreads();

    if (lane == 0) {
        float tot = 0.0f;
        for (int i = 0; i < 32; ++i) tot += partial[i];
        Ls = (int)tot;

        // ---- phase 1: peak-pick runs of probs>0.5, merge close boundaries
        int last = -1, start = -1, rmi = 0;
        float rmv = -1e30f;
        for (int j = 0; j < TDIM; ++j) {
            const float pj = pr[j];
            const bool  bj = pj > 0.5f;
            const bool  in_run = start >= 0;
            const bool  fin = (!bj) && in_run;
            if (fin) {
                int bd_idx = rmi;
                if ((bd_idx - last < MGAP) && (last > 0)) {
                    const int mid = (int)rintf((float)(bd_idx + last) * 0.5f);
                    res[last] = 0;
                    bd_idx = mid;
                }
                res[bd_idx] = 1;
                last  = bd_idx;
                start = -1;
            }
            if (bj) {
                if (!in_run) start = j;
                if (!in_run || pj > rmv) { rmv = pj; rmi = j; }
            }
        }

        // ---- phase 2: only positions with word boundary are active
        for (int j = 0; j < TDIM; ++j) {
            if (!((refb[j >> 5] >> (j & 31)) & 1u)) continue;
            int seg = 0;
            for (int d = -RGAP; d < RGAP; ++d) {
                const int w = j + d;
                if (w >= 0 && w < TDIM) seg += res[w];
            }
            const bool b0 = (seg == 0);
            const bool b1 = (seg == 1) && (res[j] != 1);
            const bool b2 = (seg > 1);
            int zidx = TDIM;
            if (b2) {
                for (int k = 1; k <= RGAP; ++k) {
                    int li = j - k; if (li < 0) li = 0;
                    int ri = j + k; if (ri > TDIM - 1) ri = TDIM - 1;
                    if (res[li] == 1 && !((refb[li >> 5] >> (li & 31)) & 1u)) { zidx = li; break; }
                    if (res[ri] == 1 && !((refb[ri >> 5] >> (ri & 31)) & 1u)) { zidx = ri; break; }
                }
            }
            if (b1) {
                for (int d = -RGAP; d < RGAP; ++d) {
                    const int w = j + d;
                    if (w >= 0 && w < TDIM)
                        res[w] = (unsigned char)((refb[w >> 5] >> (w & 31)) & 1u);
                }
            }
            if (b2 && zidx < TDIM) res[zidx] = 0;
            if (b0 || b2) res[j] = 1;
        }
    }
    __syncthreads();

    const int L = Ls;
    float* og = out_pred + (size_t)row * TDIM;
    for (int t = lane; t < TDIM; t += 32) {
        const int v = (t >= 1 && t < L - 1) ? (int)res[t] : 0;
        og[t] = (float)v;
    }
}

extern "C" void kernel_launch(void* const* d_in, const int* in_sizes, int n_in,
                              void* d_out, int out_size, void* d_ws, size_t ws_size,
                              hipStream_t stream) {
    const float* feat        = (const float*)d_in[0];
    const int*   word_bd     = (const int*)d_in[1];
    const float* non_padding = (const float*)d_in[2];
    const float* Wm          = (const float*)d_in[3];
    const float* bias        = (const float*)d_in[4];

    float* out       = (float*)d_out;
    float* out_bd    = out;                                        // B*T
    float* out_frame = out + (size_t)BDIM * TDIM;                  // B*T*104
    float* out_pred  = out_frame + (size_t)BDIM * TDIM * (CDIM-1); // B*T

    // workspace layout: probs (B*T f32) | Whi (7*8*32*16 bf16) | Wlo (same)
    char*  ws    = (char*)d_ws;
    float* probs = (float*)ws;
    __bf16* Whi  = (__bf16*)(ws + (size_t)BDIM * TDIM * sizeof(float));
    __bf16* Wlo  = Whi + 7 * 8 * 32 * 16;

    pack_W_kernel<<<56, 32, 0, stream>>>(Wm, Whi, Wlo);

    // 6000 m-tiles, 2 per wave, 8 waves per block -> 375 blocks
    gemm_logits_kernel<<<375, 256, 0, stream>>>(feat, Whi, Wlo, bias,
                                                out_bd, out_frame, probs);
    boundary_scan_kernel<<<BDIM, 32, 0, stream>>>(probs, word_bd, non_padding,
                                                  out_pred);
}
